// WeightedConformers_670014899135
// MI455X (gfx1250) — matrix operand order repaired and checked by reading, not compiled, not run on hardware
//
#include <hip/hip_runtime.h>
#include <hip/hip_bf16.h>

// ---------------------------------------------------------------------------
// Problem constants (match reference)
// ---------------------------------------------------------------------------
#define N_ATOMS 51200
#define N_EDGES 524288
#define N_MOLS  64
#define N_CONFS 20
#define MOL_SIZE 40
#define LLAYERS 3
#define DDIM    256
#define NGAUSS  64
#define CUTOFF  5.0f
#define LOG2F_  0.69314718055994531f

// ---------------------------------------------------------------------------
// WMMA types / helpers (CDNA5 wave32, 16x16x32 bf16, f32 accumulate)
// ---------------------------------------------------------------------------
typedef __attribute__((ext_vector_type(16))) __bf16 v16bf;
typedef __attribute__((ext_vector_type(8)))  float  v8f;

union Frag {
    v16bf v;
    uint4 u[2];
};

__device__ __forceinline__ v8f wmma_bf16(v16bf a, v16bf b, v8f c) {
    // 8 args: (neg_a, A, neg_b, B, c_mod, C, reuse_a, reuse_b)
    return __builtin_amdgcn_wmma_f32_16x16x32_bf16(
        false, a, false, b, (short)0, c, false, false);
}

__device__ __forceinline__ unsigned short f2bf(float f) {
    union { float f; unsigned int u; } x; x.f = f;
    unsigned int r = x.u + 0x7FFFu + ((x.u >> 16) & 1u);   // RNE
    return (unsigned short)(r >> 16);
}
__device__ __forceinline__ float bf2f(unsigned short b) {
    union { unsigned int u; float f; } x; x.u = ((unsigned int)b) << 16;
    return x.f;
}

// shifted softplus: log(1+exp(x)) - log(2), numerically stable
__device__ __forceinline__ float sspf(float x) {
    float ax = fabsf(x);
    return __logf(1.0f + __expf(-ax)) + fmaxf(x, 0.0f) - LOG2F_;
}

// ---------------------------------------------------------------------------
// Weight pack: fp32 row-major [K x N] -> bf16 packed B-fragment layout.
// Fragment (kt,nt): 32 lanes x 16 bf16 contiguous;
//   lane = ((krem/8)%2)*16 + n%16 ; element j -> k = kt*32 + (j/8)*16 + ((krem/8)%2)*8 + j%8
// ---------------------------------------------------------------------------
__global__ __launch_bounds__(256) void k_pack(const float* __restrict__ W,
                                              unsigned short* __restrict__ out,
                                              int K, int N) {
    int idx = blockIdx.x * blockDim.x + threadIdx.x;
    if (idx >= K * N) return;
    int k = idx / N, n = idx - k * N;
    int kt = k >> 5, kr = k & 31;
    int hh = (kr >> 3) & 1, j = (kr & 7) | ((kr >> 4) << 3);
    int nt = n >> 4;
    int lane = hh * 16 + (n & 15);
    size_t dst = ((size_t)(kt * (N >> 4) + nt) * 32 + lane) * 16 + j;
    out[dst] = f2bf(W[idx]);
}

// ---------------------------------------------------------------------------
// Gaussian edge basis, bf16 output  g[E x 64]
// ---------------------------------------------------------------------------
__global__ __launch_bounds__(256) void k_gauss(const float* __restrict__ nxyz,
                                               const int* __restrict__ nbr,
                                               unsigned short* __restrict__ g16) {
    int e = blockIdx.x * blockDim.x + threadIdx.x;
    if (e >= N_EDGES) return;
    int i = nbr[2 * e + 0], j = nbr[2 * e + 1];
    float dx = nxyz[i * 4 + 1] - nxyz[j * 4 + 1];
    float dy = nxyz[i * 4 + 2] - nxyz[j * 4 + 2];
    float dz = nxyz[i * 4 + 3] - nxyz[j * 4 + 3];
    float d = sqrtf(dx * dx + dy * dy + dz * dz + 1e-12f);
    const float width = CUTOFF / (float)(NGAUSS - 1);
    const float inv = 1.0f / width;
    unsigned short* gr = g16 + (size_t)e * NGAUSS;
#pragma unroll 8
    for (int k = 0; k < NGAUSS; ++k) {
        float t = (d - (float)k * width) * inv;
        gr[k] = f2bf(__expf(-0.5f * t * t));
    }
}

// ---------------------------------------------------------------------------
// Embedding gather: r = emb[z] (fp32 + bf16 copies)
// ---------------------------------------------------------------------------
__global__ __launch_bounds__(256) void k_embed(const float* __restrict__ emb,
                                               const int* __restrict__ z,
                                               float* __restrict__ r,
                                               unsigned short* __restrict__ r16) {
    int idx = blockIdx.x * blockDim.x + threadIdx.x;
    if (idx >= N_ATOMS * DDIM) return;
    int a = idx >> 8, c = idx & 255;
    float v = emb[(size_t)z[a] * DDIM + c];
    r[idx] = v;
    r16[idx] = f2bf(v);
}

__global__ __launch_bounds__(256) void k_cast(const float* __restrict__ in,
                                              unsigned short* __restrict__ out, int n) {
    int idx = blockIdx.x * blockDim.x + threadIdx.x;
    if (idx < n) out[idx] = f2bf(in[idx]);
}

// ---------------------------------------------------------------------------
// Generic WMMA GEMM: out = act(A_bf16[MxK] @ Wpacked + bias) (+ resid)
// block = 256 threads = 8 waves; each wave owns TWO adjacent 16x16 C tiles
// (A fragment loaded once, fed to two WMMAs). Block covers 16 rows x 256
// cols; grid = (M/16, ceil(N/256)).
// ---------------------------------------------------------------------------
__global__ __launch_bounds__(256) void k_gemm(const unsigned short* __restrict__ A,
                                              const unsigned short* __restrict__ Wp,
                                              const float* __restrict__ bias,
                                              const float* resid,
                                              float* outF,
                                              unsigned short* outB,
                                              int M, int K, int N, int act) {
    const int lane = threadIdx.x & 31;
    const int wave = threadIdx.x >> 5;
    const int m0 = blockIdx.x * 16;
    const int ntiles = N >> 4;
    const int nt0 = blockIdx.y * 16 + wave * 2;
    if (m0 >= M || nt0 >= ntiles) return;
    const bool has2 = (nt0 + 1) < ntiles;
    const int nt1 = has2 ? nt0 + 1 : nt0;  // clamp: boundary wave repeats tile0
    const int h = lane >> 4;
    const int mrow = lane & 15;
    const int ktiles = K >> 5;

    v8f acc0 = {}, acc1 = {};
    const unsigned short* arow = A + (size_t)(m0 + mrow) * K + h * 8;
    const unsigned short* bp0 = Wp + ((size_t)nt0 * 32 + lane) * 16;
    const unsigned short* bp1 = Wp + ((size_t)nt1 * 32 + lane) * 16;
    const size_t bstep = (size_t)ntiles * 512;
    for (int kt = 0; kt < ktiles; ++kt) {
        Frag a, b0, b1;
        const unsigned short* ap = arow + kt * 32;
        a.u[0] = *(const uint4*)(ap);
        a.u[1] = *(const uint4*)(ap + 16);
        b0.u[0] = *(const uint4*)(bp0);
        b0.u[1] = *(const uint4*)(bp0 + 8);
        b1.u[0] = *(const uint4*)(bp1);
        b1.u[1] = *(const uint4*)(bp1 + 8);
        __builtin_prefetch((const void*)(bp0 + bstep), 0, 1);  // global_prefetch_b8
        bp0 += bstep;
        bp1 += bstep;
        acc0 = wmma_bf16(a.v, b0.v, acc0);
        acc1 = wmma_bf16(a.v, b1.v, acc1);
    }

#pragma unroll
    for (int t = 0; t < 2; ++t) {
        if (t == 1 && !has2) break;
        const v8f& acc = t ? acc1 : acc0;
        const int ncol = (t ? nt1 : nt0) * 16 + mrow;
        const float bn = bias ? bias[ncol] : 0.0f;
#pragma unroll
        for (int v = 0; v < 8; ++v) {
            int m = m0 + v + h * 8;
            float x = acc[v] + bn;
            if (act) x = sspf(x);
            if (resid) x += resid[(size_t)m * N + ncol];
            if (outF) outF[(size_t)m * N + ncol] = x;
            if (outB) outB[(size_t)m * N + ncol] = f2bf(x);
        }
    }
}

// ---------------------------------------------------------------------------
// Fused edge kernel (the hot loop):
//   t1   = ssp(g_tile @ Wf1 + bf1)          (16 x 256, -> LDS bf16)
//   filt = t1 @ Wf2 + bf2                   (16 x 256)
//   msg  = phi16[a1] * filt ; atomicAdd agg[a0]
// One 16-edge tile per block iteration; 8 waves, each wave owns 2 n-tiles
// and shares one A fragment per k-step across both WMMAs.
// ---------------------------------------------------------------------------
__global__ __launch_bounds__(256) void k_edge(const unsigned short* __restrict__ g16,
                                              const unsigned short* __restrict__ phi16,
                                              const unsigned short* __restrict__ Wf1p,
                                              const float* __restrict__ bf1,
                                              const unsigned short* __restrict__ Wf2p,
                                              const float* __restrict__ bf2,
                                              const int* __restrict__ nbr,
                                              float* __restrict__ agg) {
    __shared__ __align__(16) unsigned short sh_t1[16 * 256];
    __shared__ __align__(16) unsigned short sh_phi[16 * 256];
    __shared__ int sh_a0[16];
    __shared__ int sh_a1[16];

    const int lane = threadIdx.x & 31;
    const int wave = threadIdx.x >> 5;
    const int h = lane >> 4;
    const int mrow = lane & 15;
    const int nt0 = wave * 2;         // this wave's two n-tiles
    const int nc0 = nt0 * 16 + mrow;
    const int nc1 = nc0 + 16;
    const int tilesTot = N_EDGES >> 4;

    for (int tile = blockIdx.x; tile < tilesTot; tile += gridDim.x) {
        const int e0 = tile << 4;
        __syncthreads();  // previous iteration fully consumed LDS
        if (threadIdx.x < 16) {
            sh_a0[threadIdx.x] = nbr[2 * (e0 + threadIdx.x) + 0];
            sh_a1[threadIdx.x] = nbr[2 * (e0 + threadIdx.x) + 1];
        }
        __syncthreads();

        // ---- phase A: t1 = ssp(g @ Wf1 + bf1), K = 64 (2 k-steps) ----
        {
            const unsigned short* grow = g16 + (size_t)(e0 + mrow) * NGAUSS + h * 8;
            const unsigned short* bp0 = Wf1p + ((size_t)nt0 * 32 + lane) * 16;
            const unsigned short* bp1 = Wf1p + ((size_t)(nt0 + 1) * 32 + lane) * 16;
            v8f acc0 = {}, acc1 = {};
#pragma unroll
            for (int kt = 0; kt < 2; ++kt) {
                Frag a, b0, b1;
                const unsigned short* ap = grow + kt * 32;
                a.u[0] = *(const uint4*)(ap);
                a.u[1] = *(const uint4*)(ap + 16);
                b0.u[0] = *(const uint4*)(bp0);
                b0.u[1] = *(const uint4*)(bp0 + 8);
                b1.u[0] = *(const uint4*)(bp1);
                b1.u[1] = *(const uint4*)(bp1 + 8);
                bp0 += 16 * 512;  // ntiles = 16
                bp1 += 16 * 512;
                acc0 = wmma_bf16(a.v, b0.v, acc0);
                acc1 = wmma_bf16(a.v, b1.v, acc1);
            }
            const float bn0 = bf1[nc0];
            const float bn1 = bf1[nc1];
#pragma unroll
            for (int v = 0; v < 8; ++v) {
                const int m = v + h * 8;
                sh_t1[m * 256 + nc0] = f2bf(sspf(acc0[v] + bn0));
                sh_t1[m * 256 + nc1] = f2bf(sspf(acc1[v] + bn1));
            }
        }

        // ---- gather phi rows for a1 (16 x 256 bf16 into LDS) ----
        {
            const int rrow = threadIdx.x >> 4;
            const int cc = (threadIdx.x & 15) * 16;
            const unsigned short* src = phi16 + (size_t)sh_a1[rrow] * DDIM + cc;
            uint4 p0 = *(const uint4*)(src);
            uint4 p1 = *(const uint4*)(src + 8);
            *(uint4*)&sh_phi[rrow * 256 + cc] = p0;
            *(uint4*)&sh_phi[rrow * 256 + cc + 8] = p1;
        }
        __syncthreads();

        // ---- phase B: filt = t1 @ Wf2 + bf2 ; msg ; atomic scatter ----
        {
            const unsigned short* ar = &sh_t1[mrow * 256 + h * 8];
            const unsigned short* bp0 = Wf2p + ((size_t)nt0 * 32 + lane) * 16;
            const unsigned short* bp1 = Wf2p + ((size_t)(nt0 + 1) * 32 + lane) * 16;
            v8f acc0 = {}, acc1 = {};
#pragma unroll
            for (int kt = 0; kt < 8; ++kt) {
                Frag a, b0, b1;
                const unsigned short* ap = ar + kt * 32;
                a.u[0] = *(const uint4*)(ap);
                a.u[1] = *(const uint4*)(ap + 16);
                b0.u[0] = *(const uint4*)(bp0);
                b0.u[1] = *(const uint4*)(bp0 + 8);
                b1.u[0] = *(const uint4*)(bp1);
                b1.u[1] = *(const uint4*)(bp1 + 8);
                bp0 += 16 * 512;
                bp1 += 16 * 512;
                acc0 = wmma_bf16(a.v, b0.v, acc0);
                acc1 = wmma_bf16(a.v, b1.v, acc1);
            }
            const float bn0 = bf2[nc0];
            const float bn1 = bf2[nc1];
#pragma unroll
            for (int v = 0; v < 8; ++v) {
                const int m = v + h * 8;
                const size_t arow = (size_t)sh_a0[m] * DDIM;
                float msg0 = bf2f(sh_phi[m * 256 + nc0]) * (acc0[v] + bn0);
                float msg1 = bf2f(sh_phi[m * 256 + nc1]) * (acc1[v] + bn1);
                atomicAdd(&agg[arow + nc0], msg0);
                atomicAdd(&agg[arow + nc1], msg1);
            }
        }
    }
}

// ---------------------------------------------------------------------------
// Readout helpers
// ---------------------------------------------------------------------------
__global__ __launch_bounds__(256) void k_confsum(const float* __restrict__ r,
                                                 float* __restrict__ cs) {
    const int rc = blockIdx.x;   // 0..1279 (mol*20+conf)
    const int ch = threadIdx.x;  // 0..255
    const float* base = r + ((size_t)rc * MOL_SIZE) * DDIM + ch;
    float s = 0.0f;
#pragma unroll 8
    for (int a = 0; a < MOL_SIZE; ++a) s += base[(size_t)a * DDIM];
    cs[(size_t)rc * DDIM + ch] = s;
}

__global__ __launch_bounds__(256) void k_fp(const float* __restrict__ w,
                                            const float* __restrict__ hbuf,
                                            float* __restrict__ fp) {
    int idx = blockIdx.x * blockDim.x + threadIdx.x;
    if (idx >= N_MOLS * 512) return;
    int m = idx >> 9, d = idx & 511;
    float s = 0.0f;
#pragma unroll
    for (int c = 0; c < N_CONFS; ++c)
        s += w[m * N_CONFS + c] * hbuf[((size_t)m * N_CONFS + c) * 512 + d];
    fp[idx] = s;
}

__global__ __launch_bounds__(256) void k_readout(const float* __restrict__ fp,
                                                 const float* __restrict__ rW1,
                                                 const float* __restrict__ rb1,
                                                 const float* __restrict__ rW2,
                                                 const float* __restrict__ rb2,
                                                 float* __restrict__ out) {
    __shared__ float red[256];
    const int m = blockIdx.x;
    const int j = threadIdx.x;
    const float* fr = fp + (size_t)m * 512;
    float s = rb1[j];
    for (int k = 0; k < 512; ++k) s += fr[k] * rW1[k * 256 + j];
    red[j] = sspf(s) * rW2[j];
    __syncthreads();
    for (int w = 128; w > 0; w >>= 1) {
        if (j < w) red[j] += red[j + w];
        __syncthreads();
    }
    if (j == 0) out[m] = red[0] + rb2[0];
}

// ---------------------------------------------------------------------------
// Host-side orchestration
// ---------------------------------------------------------------------------
extern "C" void kernel_launch(void* const* d_in, const int* in_sizes, int n_in,
                              void* d_out, int out_size, void* d_ws, size_t ws_size,
                              hipStream_t stream) {
    const float* nxyz   = (const float*)d_in[0];
    const float* wconf  = (const float*)d_in[1];
    const float* emb    = (const float*)d_in[2];
    const float* Wm     = (const float*)d_in[3];
    const float* bm     = (const float*)d_in[4];
    const float* Wf1    = (const float*)d_in[5];
    const float* bf1    = (const float*)d_in[6];
    const float* Wf2    = (const float*)d_in[7];
    const float* bf2    = (const float*)d_in[8];
    const float* Wo1    = (const float*)d_in[9];
    const float* bo1    = (const float*)d_in[10];
    const float* Wo2    = (const float*)d_in[11];
    const float* bo2    = (const float*)d_in[12];
    const float* mW1    = (const float*)d_in[13];
    const float* mb1    = (const float*)d_in[14];
    const float* mW2    = (const float*)d_in[15];
    const float* mb2    = (const float*)d_in[16];
    const float* rW1    = (const float*)d_in[17];
    const float* rb1    = (const float*)d_in[18];
    const float* rW2    = (const float*)d_in[19];
    const float* rb2    = (const float*)d_in[20];
    const int*   z      = (const int*)d_in[21];
    const int*   nbr    = (const int*)d_in[22];
    float* out = (float*)d_out;

    // workspace carve-up (all offsets 256B aligned)
    char* ws = (char*)d_ws;
    size_t off = 0;
    auto carve = [&](size_t bytes) { char* p = ws + off; off += (bytes + 255) & ~(size_t)255; return p; };

    unsigned short* g16   = (unsigned short*)carve((size_t)N_EDGES * NGAUSS * 2);   // 64 MB
    unsigned short* phi16 = (unsigned short*)carve((size_t)N_ATOMS * DDIM * 2);     // 25 MB (reused as agg16)
    unsigned short* t16   = (unsigned short*)carve((size_t)N_ATOMS * DDIM * 2);     // 25 MB (reused as cs16)
    float*          agg   = (float*)carve((size_t)N_ATOMS * DDIM * 4);              // 50 MB
    float*          rbuf  = (float*)carve((size_t)N_ATOMS * DDIM * 4);              // 50 MB
    unsigned short* r16   = (unsigned short*)carve((size_t)N_ATOMS * DDIM * 2);     // 25 MB
    unsigned short* Wmp   = (unsigned short*)carve((size_t)LLAYERS * 256 * 256 * 2);
    unsigned short* Wf1p  = (unsigned short*)carve((size_t)LLAYERS * 64 * 256 * 2);
    unsigned short* Wf2p  = (unsigned short*)carve((size_t)LLAYERS * 256 * 256 * 2);
    unsigned short* Wo1p  = (unsigned short*)carve((size_t)LLAYERS * 256 * 256 * 2);
    unsigned short* Wo2p  = (unsigned short*)carve((size_t)LLAYERS * 256 * 256 * 2);
    unsigned short* mW1p  = (unsigned short*)carve((size_t)256 * 384 * 2);
    unsigned short* mW2p  = (unsigned short*)carve((size_t)384 * 512 * 2);
    float*          cs    = (float*)carve((size_t)N_MOLS * N_CONFS * DDIM * 4);
    unsigned short* h1_16 = (unsigned short*)carve((size_t)N_MOLS * N_CONFS * 384 * 2);
    float*          hbuf  = (float*)carve((size_t)N_MOLS * N_CONFS * 512 * 4);
    float*          fpb   = (float*)carve((size_t)N_MOLS * 512 * 4);

    // ---- pack all weights into WMMA B-fragment layout ----
    for (int l = 0; l < LLAYERS; ++l) {
        k_pack<<<(256 * 256) / 256, 256, 0, stream>>>(Wm  + (size_t)l * 65536, Wmp  + (size_t)l * 65536, 256, 256);
        k_pack<<<(64  * 256) / 256, 256, 0, stream>>>(Wf1 + (size_t)l * 16384, Wf1p + (size_t)l * 16384, 64, 256);
        k_pack<<<(256 * 256) / 256, 256, 0, stream>>>(Wf2 + (size_t)l * 65536, Wf2p + (size_t)l * 65536, 256, 256);
        k_pack<<<(256 * 256) / 256, 256, 0, stream>>>(Wo1 + (size_t)l * 65536, Wo1p + (size_t)l * 65536, 256, 256);
        k_pack<<<(256 * 256) / 256, 256, 0, stream>>>(Wo2 + (size_t)l * 65536, Wo2p + (size_t)l * 65536, 256, 256);
    }
    k_pack<<<(256 * 384) / 256, 256, 0, stream>>>(mW1, mW1p, 256, 384);
    k_pack<<<(384 * 512) / 256, 256, 0, stream>>>(mW2, mW2p, 384, 512);

    // ---- edge gaussians + embedding gather ----
    k_gauss<<<N_EDGES / 256, 256, 0, stream>>>(nxyz, nbr, g16);
    k_embed<<<(N_ATOMS * DDIM) / 256, 256, 0, stream>>>(emb, z, rbuf, r16);

    // ---- message-passing layers ----
    const dim3 gemmGrid(N_ATOMS / 16, 1);  // N=256 -> one 256-col strip
    for (int l = 0; l < LLAYERS; ++l) {
        // phi16 = r16 @ Wm + bm
        k_gemm<<<gemmGrid, 256, 0, stream>>>(r16, Wmp + (size_t)l * 65536, bm + l * 256,
                                             nullptr, nullptr, phi16,
                                             N_ATOMS, 256, 256, 0);
        hipMemsetAsync(agg, 0, (size_t)N_ATOMS * DDIM * 4, stream);
        // fused edge filter + message + segment-sum
        k_edge<<<2048, 256, 0, stream>>>(g16, phi16,
                                         Wf1p + (size_t)l * 16384, bf1 + l * 256,
                                         Wf2p + (size_t)l * 65536, bf2 + l * 256,
                                         nbr, agg);
        // agg -> bf16 (reuse phi16 buffer)
        k_cast<<<(N_ATOMS * DDIM) / 256, 256, 0, stream>>>(agg, phi16, N_ATOMS * DDIM);
        // t16 = ssp(agg @ Wo1 + bo1)
        k_gemm<<<gemmGrid, 256, 0, stream>>>(phi16, Wo1p + (size_t)l * 65536, bo1 + l * 256,
                                             nullptr, nullptr, t16,
                                             N_ATOMS, 256, 256, 1);
        // r = r + (t16 @ Wo2 + bo2) ; refresh r16
        k_gemm<<<gemmGrid, 256, 0, stream>>>(t16, Wo2p + (size_t)l * 65536, bo2 + l * 256,
                                             rbuf, rbuf, r16,
                                             N_ATOMS, 256, 256, 0);
    }

    // ---- readout ----
    k_confsum<<<N_MOLS * N_CONFS, 256, 0, stream>>>(rbuf, cs);
    k_cast<<<(N_MOLS * N_CONFS * DDIM) / 256, 256, 0, stream>>>(cs, t16, N_MOLS * N_CONFS * DDIM);
    // h1 = ssp(cs @ mW1 + mb1)   (1280 x 384 -> 24 n-tiles -> grid.y = 2)
    k_gemm<<<dim3((N_MOLS * N_CONFS) / 16, 2), 256, 0, stream>>>(
        t16, mW1p, mb1, nullptr, nullptr, h1_16, N_MOLS * N_CONFS, 256, 384, 1);
    // h = h1 @ mW2 + mb2         (1280 x 512 -> 32 n-tiles -> grid.y = 2)
    k_gemm<<<dim3((N_MOLS * N_CONFS) / 16, 2), 256, 0, stream>>>(
        h1_16, mW2p, mb2, nullptr, hbuf, nullptr, N_MOLS * N_CONFS, 384, 512, 0);
    // fp = einsum('mc,mcd->md')
    k_fp<<<(N_MOLS * 512) / 256, 256, 0, stream>>>(wconf, hbuf, fpb);
    // out = ssp(fp @ rW1 + rb1) @ rW2 + rb2
    k_readout<<<N_MOLS, 256, 0, stream>>>(fpb, rW1, rb1, rW2, rb2, out);
}
